// _PairwiseMessages_73607149519558
// MI455X (gfx1250) — compile-verified, matrix operand order checked by reading, not run on hardware
//
#include <hip/hip_runtime.h>
#include <hip/hip_bf16.h>

// ---------------------------------------------------------------------------
// PairwiseMessages fused kernel for MI455X (gfx1250, wave32, WMMA)
//
//   out[b,q,k,:] = silu(concat(qmsg[q], kmsg[k], dot[q,k,:], dist[q,k,:]) @ W1 + b1) @ W2 + b2
//   dist_e = sqrt(|q_e|^2 + |k_e|^2 - 2*dot_e)   (avoids [B,Nq,Nk,3,64] tensor)
//
// Main kernel: 1 WG = 16q x 16k pair tile (256 pairs x 256 feats, f16 in LDS).
// 8 waves; each wave owns 2 M-tiles and runs both GEMMs on the WMMA pipe with
// dual accumulator chains so every B fragment load feeds 2 independent WMMAs.
// ---------------------------------------------------------------------------

#define BB      4
#define NQ      384
#define NK      384
#define D_EQUI  64
#define D_INV   256
#define D_MSG   64
#define IN_F    256   // 64+64+64+64
#define D_FF    256
#define D_OUT   64

#define LDSTR   264   // feats/h LDS row stride in f16 (256 + 8 pad -> bank-spread)

typedef __attribute__((ext_vector_type(16))) _Float16 v16h;
typedef __attribute__((ext_vector_type(8)))  _Float16 v8h;
typedef __attribute__((ext_vector_type(8)))  float    v8f;

// ---------------- precompute: message projections (f32 -> f16) -------------
__global__ __launch_bounds__(64) void msg_proj_kernel(
    const float* __restrict__ inv, const float* __restrict__ W,
    const float* __restrict__ bias, _Float16* __restrict__ msg)
{
  const int row = blockIdx.x;      // b*N + n
  const int j   = threadIdx.x;     // 0..63
  const float* x = inv + (size_t)row * D_INV;
  float acc = bias[j];
  for (int i = 0; i < D_INV; ++i)
    acc += x[i] * W[i * D_MSG + j];
  msg[(size_t)row * D_MSG + j] = (_Float16)acc;
}

// ---------------- precompute: Wt[n*K + k] = (f16) W[k*N + n] ---------------
__global__ __launch_bounds__(256) void wtrans_kernel(
    const float* __restrict__ W, _Float16* __restrict__ Wt, int K, int N)
{
  int idx = blockIdx.x * 256 + threadIdx.x;
  if (idx >= K * N) return;
  int k = idx % K, n = idx / K;
  Wt[idx] = (_Float16)W[k * N + n];
}

// A-style fragment loader: lane = row (&15), half-wave selects K blocks.
// elems 0..7  <- K = 32*kc + 8*hf + 0..7
// elems 8..15 <- K = 32*kc + 16 + 8*hf + 0..7
__device__ __forceinline__ v16h load_frag(const _Float16* row, int kc, int hf)
{
  union { v16h v; v8h h[2]; } u;
  u.h[0] = *(const v8h*)(row + kc * 32 + 8 * hf);
  u.h[1] = *(const v8h*)(row + kc * 32 + 16 + 8 * hf);
  return u.v;
}

__device__ __forceinline__ float silu(float x)
{
  return x / (1.0f + __expf(-x));
}

// ---------------- main fused pairwise kernel -------------------------------
__global__ __launch_bounds__(256) void pair_kernel(
    const float*    __restrict__ q_equi,  // [B,NQ,3,64]
    const float*    __restrict__ k_equi,  // [B,NK,3,64]
    const _Float16* __restrict__ qmsg,    // [B,NQ,64] f16
    const _Float16* __restrict__ kmsg,    // [B,NK,64] f16
    const _Float16* __restrict__ W1t,     // [256(ff),256(k)] f16  (= W1^T)
    const _Float16* __restrict__ W2t,     // [64(out),256(k)] f16  (= W2^T)
    const float*    __restrict__ b1,      // [256]
    const float*    __restrict__ b2,      // [64]
    float*          __restrict__ out)     // [B,NQ,NK,64]
{
  extern __shared__ __align__(16) char smem[];
  _Float16* feats = (_Float16*)smem;                        // 256*264 f16 = 135168 B (reused for h)
  float*    qe    = (float*)(smem + 135168);                // 16*3*64 f32 = 12288 B
  float*    ke    = (float*)(smem + 135168 + 12288);        // 12288 B
  _Float16* qm    = (_Float16*)(smem + 135168 + 24576);     // 16*64 f16 = 2048 B
  _Float16* km    = (_Float16*)(smem + 135168 + 26624);     // 2048 B

  const int t  = threadIdx.x;
  const int b  = blockIdx.z;
  const int q0 = blockIdx.y * 16;
  const int k0 = blockIdx.x * 16;

  // ---- stage equi + msg tiles into LDS (contiguous, coalesced) ----
  {
    const float* qsrc = q_equi + (size_t)(b * NQ + q0) * (3 * D_EQUI);
    const float* ksrc = k_equi + (size_t)(b * NK + k0) * (3 * D_EQUI);
    for (int i = t; i < 16 * 3 * D_EQUI; i += 256) { qe[i] = qsrc[i]; ke[i] = ksrc[i]; }
    const unsigned long long* s1 = (const unsigned long long*)(qmsg + (size_t)(b * NQ + q0) * D_MSG);
    const unsigned long long* s2 = (const unsigned long long*)(kmsg + (size_t)(b * NK + k0) * D_MSG);
    ((unsigned long long*)qm)[t] = s1[t];   // 256 * u64 = 1024 f16
    ((unsigned long long*)km)[t] = s2[t];
  }
  __syncthreads();

  // ---- build feature row for pair p = t : q_local = t>>4, k_local = t&15 ----
  {
    const int ql = t >> 4, kl = t & 15;
    v8h* rw = (v8h*)(feats + t * LDSTR);
    const v8h* qmr = (const v8h*)(qm + ql * D_MSG);
    const v8h* kmr = (const v8h*)(km + kl * D_MSG);
    #pragma unroll
    for (int j = 0; j < 8; ++j) { rw[j] = qmr[j]; rw[8 + j] = kmr[j]; }
    const float* qrow = qe + ql * 192;
    const float* krow = ke + kl * 192;
    for (int eb = 0; eb < 8; ++eb) {
      v8h dotv, distv;
      #pragma unroll
      for (int j = 0; j < 8; ++j) {
        const int e = eb * 8 + j;
        float a0 = qrow[e], a1 = qrow[64 + e], a2 = qrow[128 + e];
        float c0 = krow[e], c1 = krow[64 + e], c2 = krow[128 + e];
        float dp = a0 * c0 + a1 * c1 + a2 * c2;
        float qn = a0 * a0 + a1 * a1 + a2 * a2;
        float kn = c0 * c0 + c1 * c1 + c2 * c2;
        float ds = sqrtf(fmaxf(qn + kn - 2.0f * dp, 0.0f));
        dotv[j]  = (_Float16)dp;
        distv[j] = (_Float16)ds;
      }
      rw[16 + eb] = dotv;   // feats[.., 128 + 8*eb ..]
      rw[24 + eb] = distv;  // feats[.., 192 + 8*eb ..]
    }
  }
  __syncthreads();

  // ---- per-wave GEMM pipeline: wave owns m-tiles {2w, 2w+1} processed
  //      together so each B fragment feeds two independent WMMA chains ----
  const int wave = t >> 5;
  const int lane = t & 31;
  const int ln   = lane & 15;   // row (A) / col (B,C)
  const int hf   = lane >> 4;   // half-wave K-block select

  const _Float16* arow0 = feats + (wave * 32      + ln) * LDSTR;  // mt0 = 2w
  const _Float16* arow1 = feats + (wave * 32 + 16 + ln) * LDSTR;  // mt1 = 2w+1

  // preload A fragments for BOTH m-tiles (2 x 8 x v16h = 128 VGPRs)
  v16h a0[8], a1[8];
  #pragma unroll
  for (int kc = 0; kc < 8; ++kc) { a0[kc] = load_frag(arow0, kc, hf);
                                   a1[kc] = load_frag(arow1, kc, hf); }

  // ---- GEMM1: feats @ W1  (+b1, SiLU) -> h back into same LDS rows ----
  for (int nt = 0; nt < 16; ++nt) {
    v8f c0 = {}, c1 = {};
    const _Float16* brow = W1t + (size_t)(nt * 16 + ln) * IN_F;
    #pragma unroll
    for (int kc = 0; kc < 8; ++kc) {
      v16h bf = load_frag(brow, kc, hf);
      c0 = __builtin_amdgcn_wmma_f32_16x16x32_f16(
             false, a0[kc], false, bf, (short)0, c0, false, false);
      c1 = __builtin_amdgcn_wmma_f32_16x16x32_f16(
             false, a1[kc], false, bf, (short)0, c1, false, false);
    }
    const float bias = b1[nt * 16 + ln];
    _Float16* h0 = feats + (wave * 32      + 8 * hf) * LDSTR + nt * 16 + ln;
    _Float16* h1 = feats + (wave * 32 + 16 + 8 * hf) * LDSTR + nt * 16 + ln;
    #pragma unroll
    for (int r = 0; r < 8; ++r) {
      h0[r * LDSTR] = (_Float16)silu(c0[r] + bias);
      h1[r * LDSTR] = (_Float16)silu(c1[r] + bias);
    }
  }

  // ---- GEMM2: h @ W2 (+b2) -> global. Re-read A frags (now h) from LDS.
  //      Safe without barrier: this wave wrote exactly these rows. ----
  #pragma unroll
  for (int kc = 0; kc < 8; ++kc) { a0[kc] = load_frag(arow0, kc, hf);
                                   a1[kc] = load_frag(arow1, kc, hf); }

  float* ob0 = out + ((size_t)((b * NQ + q0 + wave * 2    ) * NK + k0)) * D_OUT;
  float* ob1 = out + ((size_t)((b * NQ + q0 + wave * 2 + 1) * NK + k0)) * D_OUT;
  for (int nt = 0; nt < 4; ++nt) {
    v8f c0 = {}, c1 = {};
    const _Float16* brow = W2t + (size_t)(nt * 16 + ln) * D_FF;
    #pragma unroll
    for (int kc = 0; kc < 8; ++kc) {
      v16h bf = load_frag(brow, kc, hf);
      c0 = __builtin_amdgcn_wmma_f32_16x16x32_f16(
             false, a0[kc], false, bf, (short)0, c0, false, false);
      c1 = __builtin_amdgcn_wmma_f32_16x16x32_f16(
             false, a1[kc], false, bf, (short)0, c1, false, false);
    }
    const float bias2 = b2[nt * 16 + ln];
    #pragma unroll
    for (int r = 0; r < 8; ++r) {
      const int rit = r + 8 * hf;                  // k-local row in tile
      ob0[(size_t)rit * D_OUT + nt * 16 + ln] = c0[r] + bias2;
      ob1[(size_t)rit * D_OUT + nt * 16 + ln] = c1[r] + bias2;
    }
  }
}

// ---------------------------------------------------------------------------
extern "C" void kernel_launch(void* const* d_in, const int* in_sizes, int n_in,
                              void* d_out, int out_size, void* d_ws, size_t ws_size,
                              hipStream_t stream)
{
  const float* q_equi = (const float*)d_in[0];
  const float* q_inv  = (const float*)d_in[1];
  const float* k_equi = (const float*)d_in[2];
  const float* k_inv  = (const float*)d_in[3];
  const float* Wq     = (const float*)d_in[4];
  const float* bq     = (const float*)d_in[5];
  const float* Wk     = (const float*)d_in[6];
  const float* bk     = (const float*)d_in[7];
  const float* W1     = (const float*)d_in[8];
  const float* b1     = (const float*)d_in[9];
  const float* W2     = (const float*)d_in[10];
  const float* b2     = (const float*)d_in[11];
  float* outp = (float*)d_out;

  // workspace carve-up (f16 buffers)
  char* ws = (char*)d_ws;
  _Float16* qmsg = (_Float16*)(ws);                 // B*NQ*64 f16 = 196608 B
  _Float16* kmsg = (_Float16*)(ws + 196608);        // 196608 B
  _Float16* W1t  = (_Float16*)(ws + 393216);        // 256*256 f16 = 131072 B
  _Float16* W2t  = (_Float16*)(ws + 524288);        // 64*256 f16 = 32768 B
  (void)ws_size; (void)in_sizes; (void)n_in; (void)out_size;

  // message projections
  msg_proj_kernel<<<BB * NQ, 64, 0, stream>>>(q_inv, Wq, bq, qmsg);
  msg_proj_kernel<<<BB * NK, 64, 0, stream>>>(k_inv, Wk, bk, kmsg);
  // weight transposes (f16)
  wtrans_kernel<<<(IN_F * D_FF + 255) / 256, 256, 0, stream>>>(W1, W1t, IN_F, D_FF);
  wtrans_kernel<<<(D_FF * D_OUT + 255) / 256, 256, 0, stream>>>(W2, W2t, D_FF, D_OUT);

  // fused pairwise kernel: 24 x 24 x 4 workgroups, 256 threads, 160 KB LDS
  const size_t smem = 135168 + 12288 + 12288 + 2048 + 2048;   // 163840 B
  dim3 grid(NK / 16, NQ / 16, BB);
  pair_kernel<<<grid, 256, smem, stream>>>(q_equi, k_equi, qmsg, kmsg,
                                           W1t, W2t, b1, b2, outp);
}
